// Net_7825430413946
// MI455X (gfx1250) — matrix operand order repaired and checked by reading, not compiled, not run on hardware
//
#include <hip/hip_runtime.h>
#include <hip/hip_bf16.h>
#include <math.h>

#define NN 100000
#define EE 3200000
#define FIN 256

typedef __attribute__((ext_vector_type(16))) __bf16 v16bf;
typedef __attribute__((ext_vector_type(8)))  float  v8f;

// ---------------- elementwise / graph prep kernels ----------------

__global__ __launch_bounds__(256) void zero_kernel(float* __restrict__ p, int n) {
  int i = blockIdx.x * 256 + threadIdx.x;
  if (i < n) p[i] = 0.0f;
}

__global__ __launch_bounds__(256) void deg_kernel(const int* __restrict__ dst,
                                                  float* __restrict__ deg) {
  int e = blockIdx.x * 256 + threadIdx.x;
  if (e < EE) unsafeAtomicAdd(&deg[dst[e]], 1.0f);
}

__global__ __launch_bounds__(256) void dinv_kernel(const float* __restrict__ deg,
                                                   float* __restrict__ dinv) {
  int n = blockIdx.x * 256 + threadIdx.x;
  if (n < NN) {
    float d = deg[n];
    dinv[n] = d > 0.0f ? rsqrtf(d) : 0.0f;
  }
}

__global__ __launch_bounds__(256) void norm_kernel(const int* __restrict__ src,
                                                   const int* __restrict__ dst,
                                                   const float* __restrict__ dinv,
                                                   float* __restrict__ norm) {
  int e = blockIdx.x * 256 + threadIdx.x;
  if (e < EE) norm[e] = dinv[src[e]] * dinv[dst[e]];
}

// ---------------- GEMM1: x[N,256] x W[256,64] via bf16 WMMA ----------------
// Columns 0-15: init_w1 stack0, 16-31: init_w1 stack1, 32-47: root_w1 stack0,
// 48-63: root_w1 stack1. One wave per 16-row tile; 8 K-steps of 32.

__global__ __launch_bounds__(256)
void gemm1_kernel(const float* __restrict__ x,
                  const float* __restrict__ init_w1,
                  const float* __restrict__ root_w1,
                  float* __restrict__ h1init,   // [N,32]
                  float* __restrict__ root1) {  // [N,32]
  // LDS weights, column-major: Wlds[col*256 + k], bf16, 32KB
  __shared__ __bf16 Wlds[64 * 256];
  const int t = threadIdx.x;
  {
    int col = t >> 2;              // 0..63
    int kbase = (t & 3) * 64;      // 0,64,128,192
    int o = col & 15;
    const float* wsrc;
    if (col < 16)      wsrc = init_w1 + o;            // stack0
    else if (col < 32) wsrc = init_w1 + 4096 + o;     // stack1
    else if (col < 48) wsrc = root_w1 + o;
    else               wsrc = root_w1 + 4096 + o;
    for (int i = 0; i < 64; ++i)
      Wlds[col * 256 + kbase + i] = (__bf16)wsrc[(kbase + i) * 16];
  }
  __syncthreads();

  const int wave = t >> 5;
  const int lane = t & 31;
  const int tile = blockIdx.x * 8 + wave;
  if (tile * 16 >= NN) return;
  const int upper = lane >> 4;        // K-half selector
  const int r15 = lane & 15;
  const float* xr = x + (size_t)(tile * 16 + r15) * FIN;
  const __bf16* wb = &Wlds[r15 * 256];

  v8f acc[4];
  #pragma unroll
  for (int c = 0; c < 4; ++c) acc[c] = (v8f){};

  for (int k0 = 0; k0 < FIN; k0 += 32) {
    // A fragment: lane<16 -> K {0..15}; lane>=16 -> K {16..31}
    v16bf a;
    const int alo = k0 + upper * 16;
    const int ahi = k0 + upper * 16 + 8;
    #pragma unroll
    for (int i = 0; i < 8; ++i) {
      a[i]     = (__bf16)xr[alo + i];
      a[8 + i] = (__bf16)xr[ahi + i];
    }
    const int koff = k0 + upper * 16;  // B: lanes 0-15 K=k0..k0+15, 16-31 K=+16
    #pragma unroll
    for (int c = 0; c < 4; ++c) {
      v16bf b = *(const v16bf*)(wb + c * 16 * 256 + koff);
      acc[c] = __builtin_amdgcn_wmma_f32_16x16x32_bf16(
          false, a, false, b, (short)0, acc[c], false, false);
    }
  }

  // D layout: VGPR r -> row r (lanes<16) / r+8 (lanes>=16); lane%16 = col
  #pragma unroll
  for (int r = 0; r < 8; ++r) {
    int node = tile * 16 + r + upper * 8;
    size_t base = (size_t)node * 32;
    h1init[base + r15]      = acc[0][r];
    h1init[base + 16 + r15] = acc[1][r];
    root1 [base + r15]      = acc[2][r];
    root1 [base + 16 + r15] = acc[3][r];
  }
}

// ---------------- edge scatter: agg[dst] += h[src]*norm ----------------
// 8 threads per edge, float4 each, L2-resident float atomics.

__global__ __launch_bounds__(256)
void scatter_kernel(const int* __restrict__ src, const int* __restrict__ dst,
                    const float* __restrict__ norm,
                    const float* __restrict__ hsrc,  // [N,32]
                    float* __restrict__ agg) {       // [N,32]
  int t = blockIdx.x * 256 + threadIdx.x;
  if (t >= EE * 8) return;
  int e = t >> 3, q = t & 7;
  int s = src[e], d = dst[e];
  float w = norm[e];
  const float4 v = ((const float4*)hsrc)[(size_t)s * 8 + q];
  float* ap = agg + (size_t)d * 32 + q * 4;
  unsafeAtomicAdd(ap + 0, v.x * w);
  unsafeAtomicAdd(ap + 1, v.y * w);
  unsafeAtomicAdd(ap + 2, v.z * w);
  unsafeAtomicAdd(ap + 3, v.w * w);
}

// ---------------- post-conv1: relu(mean_k relu(agg+root+bias)) ----------------

__global__ __launch_bounds__(256)
void post1_kernel(const float* __restrict__ agg1, const float* __restrict__ root1,
                  const float* __restrict__ bias1,
                  float* __restrict__ h, float* __restrict__ out_feat) {
  int t = blockIdx.x * 256 + threadIdx.x;
  if (t >= NN * 16) return;
  int n = t >> 4, j = t & 15;
  size_t b = (size_t)n * 32;
  float v0 = fmaxf(agg1[b + j]      + root1[b + j]      + bias1[j],      0.0f);
  float v1 = fmaxf(agg1[b + 16 + j] + root1[b + 16 + j] + bias1[16 + j], 0.0f);
  float v = fmaxf(0.5f * (v0 + v1), 0.0f);
  h[t] = v;
  out_feat[t] = v;
}

// ---------------- GEMM2: h[N,16] x W2[16,64], K padded to 32, WMMA ----------------

__global__ __launch_bounds__(256)
void gemm2_kernel(const float* __restrict__ h,
                  const float* __restrict__ init_w2,
                  const float* __restrict__ root_w2,
                  float* __restrict__ h2init, float* __restrict__ root2) {
  __shared__ __bf16 W2[64 * 16];  // [col][k]
  int t = threadIdx.x;
  #pragma unroll
  for (int j = 0; j < 4; ++j) {
    int idx = t * 4 + j;
    int col = idx >> 4, k = idx & 15, o = col & 15;
    float v;
    if (col < 16)      v = init_w2[k * 16 + o];
    else if (col < 32) v = init_w2[256 + k * 16 + o];
    else if (col < 48) v = root_w2[k * 16 + o];
    else               v = root_w2[256 + k * 16 + o];
    W2[col * 16 + k] = (__bf16)v;
  }
  __syncthreads();

  int wave = t >> 5, lane = t & 31;
  int tile = blockIdx.x * 8 + wave;
  if (tile * 16 >= NN) return;
  int upper = lane >> 4;
  int r15 = lane & 15;
  int row = tile * 16 + r15;

  v16bf a;
  #pragma unroll
  for (int i = 0; i < 8; ++i) {
    a[i]     = upper ? (__bf16)0.0f : (__bf16)h[(size_t)row * 16 + i];      // lanes<16: K 0..7 real
    a[8 + i] = upper ? (__bf16)0.0f : (__bf16)h[(size_t)row * 16 + 8 + i];  // lanes<16: K 8..15 real; lanes>=16 K 16..31 pad
  }

  #pragma unroll
  for (int c = 0; c < 4; ++c) {
    v16bf b;
    #pragma unroll
    for (int i = 0; i < 16; ++i) b[i] = (__bf16)0.0f;
    if (lane < 16)  // lanes>=16 hold K=16..31 -> zero pad
      b = *(const v16bf*)&W2[(c * 16 + r15) * 16];
    v8f acc = (v8f){};
    acc = __builtin_amdgcn_wmma_f32_16x16x32_bf16(
        false, a, false, b, (short)0, acc, false, false);
    float* dp = (c < 2) ? h2init : root2;
    int half = (c & 1) * 16;
    #pragma unroll
    for (int r = 0; r < 8; ++r) {
      int node = tile * 16 + r + upper * 8;
      dp[(size_t)node * 32 + half + r15] = acc[r];
    }
  }
}

// ---------------- post-conv2: mean over K + log_softmax ----------------

__global__ __launch_bounds__(256)
void post2_kernel(const float* __restrict__ agg2, const float* __restrict__ root2,
                  const float* __restrict__ bias2, float* __restrict__ out) {
  int n = blockIdx.x * 256 + threadIdx.x;
  if (n >= NN) return;
  size_t b = (size_t)n * 32;
  float l[16];
  float m = -INFINITY;
  #pragma unroll
  for (int j = 0; j < 16; ++j) {
    float v0 = agg2[b + j]      + root2[b + j]      + bias2[j];
    float v1 = agg2[b + 16 + j] + root2[b + 16 + j] + bias2[16 + j];
    float v = 0.5f * (v0 + v1);
    l[j] = v;
    m = fmaxf(m, v);
  }
  float s = 0.0f;
  #pragma unroll
  for (int j = 0; j < 16; ++j) s += expf(l[j] - m);
  float lse = logf(s) + m;
  #pragma unroll
  for (int j = 0; j < 16; ++j) out[(size_t)n * 16 + j] = l[j] - lse;
}

// ---------------- host launcher ----------------

extern "C" void kernel_launch(void* const* d_in, const int* in_sizes, int n_in,
                              void* d_out, int out_size, void* d_ws, size_t ws_size,
                              hipStream_t stream) {
  const float* x       = (const float*)d_in[0];
  const int*   edge    = (const int*)d_in[1];
  const float* init_w1 = (const float*)d_in[2];
  const float* root_w1 = (const float*)d_in[3];
  const float* bias1   = (const float*)d_in[4];
  const float* init_w2 = (const float*)d_in[5];
  const float* root_w2 = (const float*)d_in[6];
  const float* bias2   = (const float*)d_in[7];
  const int* src = edge;
  const int* dst = edge + EE;

  float* w = (float*)d_ws;
  float* deg   = w;                     // N      (contiguous zero block start)
  float* agg1  = deg   + NN;            // 32N
  float* agg2  = agg1  + 32 * NN;       // 32N    (zero block = 65N floats)
  float* dinv  = agg2  + 32 * NN;       // N
  float* nrm   = dinv  + NN;            // E
  float* h1in  = nrm   + EE;            // 32N
  float* root1 = h1in  + 32 * NN;       // 32N
  float* hbuf  = root1 + 32 * NN;       // 16N
  float* h2in  = hbuf  + 16 * NN;       // 32N
  float* root2 = h2in  + 32 * NN;       // 32N

  float* out_ls = (float*)d_out;        // log_softmax [N,16]
  float* out_h  = out_ls + 16 * NN;     // agg_feature [N,16]

  const int nzero = 65 * NN;
  zero_kernel<<<(nzero + 255) / 256, 256, 0, stream>>>(w, nzero);
  deg_kernel<<<(EE + 255) / 256, 256, 0, stream>>>(dst, deg);
  dinv_kernel<<<(NN + 255) / 256, 256, 0, stream>>>(deg, dinv);
  norm_kernel<<<(EE + 255) / 256, 256, 0, stream>>>(src, dst, dinv, nrm);

  const int tiles = NN / 16;                       // 6250
  const int gblocks = (tiles + 7) / 8;             // 782
  gemm1_kernel<<<gblocks, 256, 0, stream>>>(x, init_w1, root_w1, h1in, root1);
  scatter_kernel<<<(EE * 8 + 255) / 256, 256, 0, stream>>>(src, dst, nrm, h1in, agg1);
  post1_kernel<<<(NN * 16 + 255) / 256, 256, 0, stream>>>(agg1, root1, bias1, hbuf, out_h);

  gemm2_kernel<<<gblocks, 256, 0, stream>>>(hbuf, init_w2, root_w2, h2in, root2);
  scatter_kernel<<<(EE * 8 + 255) / 256, 256, 0, stream>>>(src, dst, nrm, h2in, agg2);
  post2_kernel<<<(NN + 255) / 256, 256, 0, stream>>>(agg2, root2, bias2, out_ls);
}
